// AxisNetwork_4406636446000
// MI455X (gfx1250) — matrix operand order repaired and checked by reading, not compiled, not run on hardware
//
#include <hip/hip_runtime.h>
#include <hip/hip_bf16.h>

typedef _Float16 h2v  __attribute__((ext_vector_type(2)));
typedef _Float16 v8h  __attribute__((ext_vector_type(8)));
typedef _Float16 v16h __attribute__((ext_vector_type(16)));
typedef float    v8f  __attribute__((ext_vector_type(8)));

#define RES 512
#define ED  256
#define HID 128
#define W0F 30.0f
#define WAVES_PER_BLOCK 2
#define TILES_PER_WAVE 2
#define PTS_PER_WAVE (16 * TILES_PER_WAVE)   // 32
#define PTS_PER_BLOCK (WAVES_PER_BLOCK * PTS_PER_WAVE)  // 64

// ---------------------------------------------------------------------------
// Prep: convert f32 -> f16 (weights + embedding tables) into workspace.
// ---------------------------------------------------------------------------
__global__ void cvt_f32_to_f16(const float* __restrict__ src,
                               _Float16* __restrict__ dst, int n) {
    int i = blockIdx.x * blockDim.x + threadIdx.x;
    if (i < n) dst[i] = (_Float16)src[i];
}

// ---------------------------------------------------------------------------
// Main fused kernel: gather/interp -> WMMA layer0 -> WMMA layer1 -> layer2
// Each wave owns two 16-point tiles that share B (weight) fragments, so each
// global B load feeds two v_wmma. 2 waves (64 threads) per block, 48KB LDS.
// ---------------------------------------------------------------------------
__global__ __launch_bounds__(WAVES_PER_BLOCK * 32) void axisnet_kernel(
    const float*    __restrict__ coords,
    const _Float16* __restrict__ emb0h,
    const _Float16* __restrict__ emb1h,
    const _Float16* __restrict__ w0h,   // [HID, ED]  (w0.T[k,n] = w0h[n*ED+k])
    const float*    __restrict__ b0,
    const _Float16* __restrict__ w1h,   // [HID, HID]
    const float*    __restrict__ b1,
    const float*    __restrict__ w2,    // [3, HID] f32
    const float*    __restrict__ b2,
    float*          __restrict__ out,
    int npts)
{
    // per-wave, per-tile staging
    __shared__ alignas(32) _Float16 esm[WAVES_PER_BLOCK][TILES_PER_WAVE][16 * ED];  // 2*2*8KB = 32KB
    __shared__ alignas(32) _Float16 hsm[WAVES_PER_BLOCK][TILES_PER_WAVE][16 * HID]; // 2*2*4KB = 16KB

    const int lane = threadIdx.x & 31;
    const int wave = threadIdx.x >> 5;
    const int p0   = (blockIdx.x * WAVES_PER_BLOCK + wave) * PTS_PER_WAVE;

    // ---------------- Stage 1: per-axis linear interp, product -> e tiles ---
    for (int t = 0; t < TILES_PER_WAVE; ++t) {
        _Float16* etile = esm[wave][t];
        for (int m = 0; m < 16; ++m) {
            const int p = p0 + t * 16 + m;
            float cx = coords[2 * p + 0];
            float cy = coords[2 * p + 1];
            cx = fminf(fmaxf(cx, -1.0f), 0.999f);
            cy = fminf(fmaxf(cy, -1.0f), 0.999f);
            const float ax = (0.5f * cx + 0.5f) * (float)(RES - 1);
            const float ay = (0.5f * cy + 0.5f) * (float)(RES - 1);
            const int ix = (int)floorf(ax);
            const int iy = (int)floorf(ay);
            const float wx = ax - (float)ix;
            const float wy = ay - (float)iy;

            const h2v* r0a = (const h2v*)(emb0h + (size_t)ix * ED);
            const h2v* r0b = (const h2v*)(emb0h + (size_t)(ix + 1) * ED);
            const h2v* r1a = (const h2v*)(emb1h + (size_t)iy * ED);
            const h2v* r1b = (const h2v*)(emb1h + (size_t)(iy + 1) * ED);
            h2v* erow = (h2v*)(etile + m * ED);

            #pragma unroll
            for (int dv = lane; dv < ED / 2; dv += 32) {
                h2v a0 = r0a[dv], a1 = r0b[dv];
                h2v c0 = r1a[dv], c1 = r1b[dv];
                float ex0 = (1.0f - wx) * (float)a0.x + wx * (float)a1.x;
                float ex1 = (1.0f - wx) * (float)a0.y + wx * (float)a1.y;
                float ey0 = (1.0f - wy) * (float)c0.x + wy * (float)c1.x;
                float ey1 = (1.0f - wy) * (float)c0.y + wy * (float)c1.y;
                h2v e;
                e.x = (_Float16)(ex0 * ey0);
                e.y = (_Float16)(ex1 * ey1);
                erow[dv] = e;
            }
        }
    }
    __syncthreads();

    // Fragment lane geometry (wave32 WMMA, 16x16x32 f16)
    const int mrow   = lane & 15;             // A row / D col
    const int khalfA = (lane >> 4) << 4;      // A: +0 or +16 within K-slice
    const int khalfB = (lane >> 4) << 4;      // B: +0 or +16 within K-slice
    const int moff   = (lane >> 4) << 3;      // D rows: +0 or +8

    // ---------------- Layer 0: h = sin(30*(e @ w0.T + b0)), ED->HID --------
    {
        const _Float16* e0 = esm[wave][0];
        const _Float16* e1 = esm[wave][1];
        for (int nt = 0; nt < HID / 16; ++nt) {
            const int ncol = nt * 16 + mrow;
            v8f acc0 = {};
            v8f acc1 = {};
            #pragma unroll
            for (int ks = 0; ks < ED / 32; ++ks) {
                const int kb = ks * 32;
                // shared B fragment (weights)
                const _Float16* bp = w0h + (size_t)ncol * ED + kb + khalfB;
                v16h b = *(const v16h*)bp;
                // A fragment, tile 0
                const _Float16* ap0 = e0 + mrow * ED + kb + khalfA;
                v8h a0lo = *(const v8h*)(ap0);
                v8h a0hi = *(const v8h*)(ap0 + 8);
                v16h a0 = __builtin_shufflevector(a0lo, a0hi,
                    0, 1, 2, 3, 4, 5, 6, 7, 8, 9, 10, 11, 12, 13, 14, 15);
                // A fragment, tile 1
                const _Float16* ap1 = e1 + mrow * ED + kb + khalfA;
                v8h a1lo = *(const v8h*)(ap1);
                v8h a1hi = *(const v8h*)(ap1 + 8);
                v16h a1 = __builtin_shufflevector(a1lo, a1hi,
                    0, 1, 2, 3, 4, 5, 6, 7, 8, 9, 10, 11, 12, 13, 14, 15);
                acc0 = __builtin_amdgcn_wmma_f32_16x16x32_f16(
                    false, a0, false, b, (short)0, acc0, false, false);
                acc1 = __builtin_amdgcn_wmma_f32_16x16x32_f16(
                    false, a1, false, b, (short)0, acc1, false, false);
            }
            const float bias = b0[ncol];
            _Float16* h0 = hsm[wave][0];
            _Float16* h1 = hsm[wave][1];
            #pragma unroll
            for (int r = 0; r < 8; ++r) {
                h0[(r + moff) * HID + ncol] = (_Float16)__sinf(W0F * (acc0[r] + bias));
                h1[(r + moff) * HID + ncol] = (_Float16)__sinf(W0F * (acc1[r] + bias));
            }
        }
    }
    __syncthreads();

    // ---------------- Layer 1: h2 = sin(30*(h @ w1.T + b1)), HID->HID ------
    // write results into the (now free) e regions
    {
        const _Float16* h0 = hsm[wave][0];
        const _Float16* h1 = hsm[wave][1];
        _Float16* o0 = esm[wave][0];
        _Float16* o1 = esm[wave][1];
        for (int nt = 0; nt < HID / 16; ++nt) {
            const int ncol = nt * 16 + mrow;
            v8f acc0 = {};
            v8f acc1 = {};
            #pragma unroll
            for (int ks = 0; ks < HID / 32; ++ks) {
                const int kb = ks * 32;
                const _Float16* bp = w1h + (size_t)ncol * HID + kb + khalfB;
                v16h b = *(const v16h*)bp;
                const _Float16* ap0 = h0 + mrow * HID + kb + khalfA;
                v8h a0lo = *(const v8h*)(ap0);
                v8h a0hi = *(const v8h*)(ap0 + 8);
                v16h a0 = __builtin_shufflevector(a0lo, a0hi,
                    0, 1, 2, 3, 4, 5, 6, 7, 8, 9, 10, 11, 12, 13, 14, 15);
                const _Float16* ap1 = h1 + mrow * HID + kb + khalfA;
                v8h a1lo = *(const v8h*)(ap1);
                v8h a1hi = *(const v8h*)(ap1 + 8);
                v16h a1 = __builtin_shufflevector(a1lo, a1hi,
                    0, 1, 2, 3, 4, 5, 6, 7, 8, 9, 10, 11, 12, 13, 14, 15);
                acc0 = __builtin_amdgcn_wmma_f32_16x16x32_f16(
                    false, a0, false, b, (short)0, acc0, false, false);
                acc1 = __builtin_amdgcn_wmma_f32_16x16x32_f16(
                    false, a1, false, b, (short)0, acc1, false, false);
            }
            const float bias = b1[ncol];
            #pragma unroll
            for (int r = 0; r < 8; ++r) {
                o0[(r + moff) * HID + ncol] = (_Float16)__sinf(W0F * (acc0[r] + bias));
                o1[(r + moff) * HID + ncol] = (_Float16)__sinf(W0F * (acc1[r] + bias));
            }
        }
    }
    __syncthreads();

    // ---------------- Layer 2: out = h2 @ w2.T + b2, HID->3 (VALU) ---------
    // all 32 lanes active: lane>>4 picks the tile, lane&15 the point
    {
        const int t = lane >> 4;
        const int m = lane & 15;
        const _Float16* hrow = esm[wave][t] + m * HID;
        float o0 = b2[0], o1 = b2[1], o2 = b2[2];
        #pragma unroll 8
        for (int k = 0; k < HID; ++k) {
            const float hv = (float)hrow[k];
            o0 = fmaf(hv, w2[0 * HID + k], o0);
            o1 = fmaf(hv, w2[1 * HID + k], o1);
            o2 = fmaf(hv, w2[2 * HID + k], o2);
        }
        const int p = p0 + t * 16 + m;
        out[p * 3 + 0] = o0;
        out[p * 3 + 1] = o1;
        out[p * 3 + 2] = o2;
    }
}

// ---------------------------------------------------------------------------
extern "C" void kernel_launch(void* const* d_in, const int* in_sizes, int n_in,
                              void* d_out, int out_size, void* d_ws, size_t ws_size,
                              hipStream_t stream) {
    const float* coords = (const float*)d_in[0];
    const float* emb0   = (const float*)d_in[1];
    const float* emb1   = (const float*)d_in[2];
    const float* w0     = (const float*)d_in[3];
    const float* b0     = (const float*)d_in[4];
    const float* w1     = (const float*)d_in[5];
    const float* b1     = (const float*)d_in[6];
    const float* w2     = (const float*)d_in[7];
    const float* b2     = (const float*)d_in[8];
    float* out = (float*)d_out;

    const int npts = in_sizes[0] / 2;

    // Workspace layout (f16 copies): w0h | w1h | emb0h | emb1h  (~608KB)
    _Float16* w0h   = (_Float16*)d_ws;                 // 128*256
    _Float16* w1h   = w0h + HID * ED;                  // 128*128
    _Float16* emb0h = w1h + HID * HID;                 // 512*256
    _Float16* emb1h = emb0h + RES * ED;                // 512*256

    cvt_f32_to_f16<<<(HID * ED + 255) / 256, 256, 0, stream>>>(w0, w0h, HID * ED);
    cvt_f32_to_f16<<<(HID * HID + 255) / 256, 256, 0, stream>>>(w1, w1h, HID * HID);
    cvt_f32_to_f16<<<(RES * ED + 255) / 256, 256, 0, stream>>>(emb0, emb0h, RES * ED);
    cvt_f32_to_f16<<<(RES * ED + 255) / 256, 256, 0, stream>>>(emb1, emb1h, RES * ED);

    const int nblocks = npts / PTS_PER_BLOCK;          // 16384 for N=1M
    axisnet_kernel<<<nblocks, WAVES_PER_BLOCK * 32, 0, stream>>>(
        coords, emb0h, emb1h, w0h, b0, w1h, b1, w2, b2, out, npts);
}
